// MambaBlock_64209761075798
// MI455X (gfx1250) — matrix operand order repaired and checked by reading, not compiled, not run on hardware
//
#include <hip/hip_runtime.h>
#include <hip/hip_bf16.h>
#include <math.h>

typedef __attribute__((ext_vector_type(16))) __bf16 bf16x16;
typedef __attribute__((ext_vector_type(8)))  __bf16 bf16x8;
typedef __attribute__((ext_vector_type(8)))  float  f32x8;

#define BATCH  8
#define TLEN   2048
#define DMODEL 1024
#define DSTATE 256
#define BT     (BATCH * TLEN)   // 16384

static inline int ceil_div(int a, int b) { return (a + b - 1) / b; }

// ---------------------------------------------------------------------------
// fast tanh / sigmoid: native V_TANH_F32 on gfx1250 when available.
// ---------------------------------------------------------------------------
__device__ __forceinline__ float fast_tanh(float x) {
#if defined(__AMDGCN__) && __has_builtin(__builtin_amdgcn_tanhf)
  return __builtin_amdgcn_tanhf(x);
#elif defined(__AMDGCN__) && __has_builtin(__builtin_amdgcn_tanh_f32)
  return __builtin_amdgcn_tanh_f32(x);
#else
  float xc = fminf(fmaxf(x, -15.0f), 15.0f);
  float e  = __expf(2.0f * xc);
#if defined(__AMDGCN__) && __has_builtin(__builtin_amdgcn_rcpf)
  float r = __builtin_amdgcn_rcpf(e + 1.0f);
#else
  float r = 1.0f / (e + 1.0f);
#endif
  return 1.0f - 2.0f * r;
#endif
}

__device__ __forceinline__ float fast_sigmoid(float x) {
  float xc = fminf(fmaxf(x, -30.0f), 30.0f);
#if defined(__AMDGCN__)
  float e = __expf(-xc);
#else
  float e = expf(-xc);
#endif
#if defined(__AMDGCN__) && __has_builtin(__builtin_amdgcn_rcpf)
  return __builtin_amdgcn_rcpf(1.0f + e);
#else
  return 1.0f / (1.0f + e);
#endif
}

// LDS-only split barrier: fences DS ops but NOT global stores (keeps the
// per-step HBM writes of h_t off the serial critical path).
__device__ __forceinline__ void lds_barrier() {
#if defined(__AMDGCN__)
  asm volatile("s_wait_dscnt 0x0\n\t"
               "s_barrier_signal -1\n\t"
               "s_barrier_wait -1" ::: "memory");
#endif
}

__device__ __forceinline__ void sched_fence() {
#if defined(__AMDGCN__) && __has_builtin(__builtin_amdgcn_sched_barrier)
  __builtin_amdgcn_sched_barrier(0);
#endif
}

// ---------------------------------------------------------------------------
// Conversion kernels: f32 -> bf16, optionally transposed so every GEMM "B"
// matrix is stored row-major [N][K] (natural CDNA5 B-operand layout).
// ---------------------------------------------------------------------------
__global__ void cvt_f32_to_bf16(const float* __restrict__ src,
                                __bf16* __restrict__ dst, int n) {
  int i = blockIdx.x * blockDim.x + threadIdx.x;
  if (i < n) dst[i] = (__bf16)src[i];
}

// src [rows][cols] -> dst [cols][rows]
__global__ void cvt_transpose_bf16(const float* __restrict__ src,
                                   __bf16* __restrict__ dst, int rows, int cols) {
  int i = blockIdx.x * blockDim.x + threadIdx.x;
  if (i < rows * cols) {
    int r = i / cols, c = i % cols;
    dst[(size_t)c * rows + r] = (__bf16)src[i];
  }
}

// ---------------------------------------------------------------------------
// bf16 WMMA GEMM:  out[M,N] = epilogue( A[M,K](bf16) @ Bt[N,K](bf16)^T )
// Block tile 128x128, K-tile 32.  128 threads = 4 waves (2x2), each wave
// computes a 64x64 macro-tile (4x4 WMMA tiles): 8 fragment loads feed 16
// WMMAs per K-tile (2x better LDS:compute ratio than 64x32 waves).
// Double-buffered LDS, software pipelined; one barrier per K-tile.
// ---------------------------------------------------------------------------
#define LDSA 56   // padded bf16 row stride: 112B, 16B-aligned rows

template <bool BIAS, bool SIG, bool OUT_BF16, bool RES>
__global__ __launch_bounds__(128) void gemm_bf16_wmma(
    const __bf16* __restrict__ Ag, const __bf16* __restrict__ Btg,
    const float* __restrict__ bias, const float* __restrict__ resid,
    void* __restrict__ outp, int M, int N, int K) {
  __shared__ __bf16 As[2][128 * LDSA];
  __shared__ __bf16 Bs[2][128 * LDSA];

  const int tid   = threadIdx.x;
  const int lane  = tid & 31;
  const int wave  = tid >> 5;     // 0..3
  const int wm    = wave >> 1;    // 0..1
  const int wn    = wave & 1;     // 0..1
  const int bm    = blockIdx.x * 128;
  const int bn    = blockIdx.y * 128;
  const int l16   = lane & 15;
  const int lhalf = lane >> 4;    // 0 or 1

  // staging coordinates: four 16B chunks per thread per matrix per K-tile
  int srow[4], scol[4];
#pragma unroll
  for (int c0 = 0; c0 < 4; ++c0) {
    int c = tid + 128 * c0;       // 0..511
    srow[c0] = c >> 2;            // 0..127
    scol[c0] = (c & 3) << 3;      // 0,8,16,24
  }

  f32x8 acc[4][4];
#pragma unroll
  for (int i = 0; i < 4; ++i)
#pragma unroll
    for (int j = 0; j < 4; ++j) acc[i][j] = {};

  const int NK = K >> 5;
  bf16x8 ra[4], rb[4];

  // prologue: stage tile 0
#pragma unroll
  for (int c0 = 0; c0 < 4; ++c0) {
    ra[c0] = *(const bf16x8*)&Ag[(size_t)(bm + srow[c0]) * K + scol[c0]];
    rb[c0] = *(const bf16x8*)&Btg[(size_t)(bn + srow[c0]) * K + scol[c0]];
  }
#pragma unroll
  for (int c0 = 0; c0 < 4; ++c0) {
    *(bf16x8*)&As[0][srow[c0] * LDSA + scol[c0]] = ra[c0];
    *(bf16x8*)&Bs[0][srow[c0] * LDSA + scol[c0]] = rb[c0];
  }
  __syncthreads();

  for (int kt = 0; kt < NK; ++kt) {
    const int p = kt & 1;
    // issue next tile's global loads (consumed only after compute)
    if (kt + 1 < NK) {
      const int kk = (kt + 1) << 5;
#pragma unroll
      for (int c0 = 0; c0 < 4; ++c0) {
        ra[c0] = *(const bf16x8*)&Ag[(size_t)(bm + srow[c0]) * K + kk + scol[c0]];
        rb[c0] = *(const bf16x8*)&Btg[(size_t)(bn + srow[c0]) * K + kk + scol[c0]];
      }
    }

    // B fragments (32x16): per lane-half 16 contiguous K values
    bf16x16 bfv[4];
#pragma unroll
    for (int j = 0; j < 4; ++j) {
      const __bf16* bp = &Bs[p][(wn * 64 + j * 16 + l16) * LDSA + lhalf * 16];
      *(bf16x8*)&bfv[j]       = *(const bf16x8*)bp;
      *((bf16x8*)&bfv[j] + 1) = *(const bf16x8*)(bp + 8);
    }
#pragma unroll
    for (int i = 0; i < 4; ++i) {
      // A fragment (16x32): per lane-half 16 contiguous K values
      bf16x16 af;
      const __bf16* ap = &As[p][(wm * 64 + i * 16 + l16) * LDSA + lhalf * 16];
      *(bf16x8*)&af       = *(const bf16x8*)ap;
      *((bf16x8*)&af + 1) = *(const bf16x8*)(ap + 8);
#pragma unroll
      for (int j = 0; j < 4; ++j) {
        acc[i][j] = __builtin_amdgcn_wmma_f32_16x16x32_bf16(
            false, af, false, bfv[j], (short)0, acc[i][j], false, false);
      }
    }

    // stage next tile into the other buffer (loadcnt wait lands here)
    if (kt + 1 < NK) {
#pragma unroll
      for (int c0 = 0; c0 < 4; ++c0) {
        *(bf16x8*)&As[p ^ 1][srow[c0] * LDSA + scol[c0]] = ra[c0];
        *(bf16x8*)&Bs[p ^ 1][srow[c0] * LDSA + scol[c0]] = rb[c0];
      }
    }
    __syncthreads();
  }

  // epilogue
#pragma unroll
  for (int i = 0; i < 4; ++i) {
#pragma unroll
    for (int j = 0; j < 4; ++j) {
      int n    = bn + wn * 64 + j * 16 + l16;
      float bv = BIAS ? bias[n] : 0.0f;
#pragma unroll
      for (int r = 0; r < 8; ++r) {
        int m   = bm + wm * 64 + i * 16 + r + lhalf * 8;
        float v = acc[i][j][r] + bv;
        if (SIG) v = fast_sigmoid(v);
        if (RES) v += resid[(size_t)m * N + n];
        if (OUT_BF16)
          ((__bf16*)outp)[(size_t)m * N + n] = (__bf16)v;
        else
          ((float*)outp)[(size_t)m * N + n] = v;
      }
    }
  }
}

// ---------------------------------------------------------------------------
// Sequential recurrence:  h_t = tanh(h_{t-1} @ A + ub_t),  h [8,256] padded
// to [16,256].  16 waves; wave w owns output columns 16w..16w+15 and keeps
// A's column slab resident in registers (8 B-operand fragments).  h is
// double-buffered in LDS (t-loop unrolled by 2 -> constant buffer pointers),
// ONE LDS-only barrier per step.  All 16 ds_load_b128 for h's A-fragments
// are forced in flight before the WMMA chain (sched_barrier); ub global
// loads issue first and are consumed after the chain; two accumulator
// chains halve the WMMA D->C dependency depth.
// ---------------------------------------------------------------------------
#define HS 264  // padded bf16 row stride for h in LDS (528B, 16B-aligned)

__global__ __launch_bounds__(512) void scan_kernel(
    const __bf16* __restrict__ AT,      // A^T, [N=256][K=256] bf16
    const float* __restrict__ ubm,      // u@Bm, [B*T, 256] f32, row = b*T + t
    __bf16* __restrict__ h_all) {       // out: [T, 8, 256] bf16
  __shared__ __bf16 hsh0[16 * HS];
  __shared__ __bf16 hsh1[16 * HS];

  const int tid   = threadIdx.x;
  const int lane  = tid & 31;
  const int wave  = tid >> 5;   // 0..15
  const int l16   = lane & 15;
  const int lhalf = lane >> 4;
  const int n     = wave * 16 + l16;

  // Resident B-operands: columns [16w..16w+15] of A, all K=256 (8 frags)
  bf16x16 bfrag[8];
#pragma unroll
  for (int kb = 0; kb < 8; ++kb) {
    const __bf16* bp = &AT[(size_t)n * 256 + kb * 32 + lhalf * 16];
    *(bf16x8*)&bfrag[kb]       = *(const bf16x8*)bp;
    *((bf16x8*)&bfrag[kb] + 1) = *(const bf16x8*)(bp + 8);
  }

  // per-row ub base pointers (row index clamped: padding rows re-read row&7)
  const float* ubp[8];
#pragma unroll
  for (int r = 0; r < 8; ++r) {
    int mc = (r + lhalf * 8) & 7;
    ubp[r] = ubm + (size_t)mc * TLEN * DSTATE + n;
  }

  for (int i = tid; i < 16 * HS; i += 512) {
    hsh0[i] = (__bf16)0.0f;
    hsh1[i] = (__bf16)0.0f;
  }
  __syncthreads();

  auto step = [&](int t, const __bf16* __restrict__ src,
                  __bf16* __restrict__ dst) {
    // 1) issue ub loads (L2-resident), consumed after the WMMA chain
    float ubv[8];
#pragma unroll
    for (int r = 0; r < 8; ++r) ubv[r] = ubp[r][(size_t)t * DSTATE];

    // 2) all 16 ds_load_b128 for h_{t-1}'s A-fragments in flight
    bf16x16 af[8];
#pragma unroll
    for (int kb = 0; kb < 8; ++kb) {
      const __bf16* ap = &src[l16 * HS + kb * 32 + lhalf * 16];
      *(bf16x8*)&af[kb]       = *(const bf16x8*)ap;
      *((bf16x8*)&af[kb] + 1) = *(const bf16x8*)(ap + 8);
    }
    sched_fence();  // don't interleave loads into the WMMA chain

    // 3) h_{t-1} @ A : two interleaved WMMA chains (depth 4 each)
    f32x8 acc0 = {}, acc1 = {};
#pragma unroll
    for (int kb = 0; kb < 8; kb += 2) {
      acc0 = __builtin_amdgcn_wmma_f32_16x16x32_bf16(
          false, af[kb], false, bfrag[kb], (short)0, acc0, false, false);
      acc1 = __builtin_amdgcn_wmma_f32_16x16x32_bf16(
          false, af[kb + 1], false, bfrag[kb + 1], (short)0, acc1, false, false);
    }

    // 4) combine + tanh, write h_t (LDS ping-pong + background global store)
    __bf16 hv[8];
#pragma unroll
    for (int r = 0; r < 8; ++r)
      hv[r] = (__bf16)fast_tanh(acc0[r] + acc1[r] + ubv[r]);
#pragma unroll
    for (int r = 0; r < 8; ++r)
      dst[(r + lhalf * 8) * HS + n] = hv[r];
    if (lhalf == 0) {
#pragma unroll
      for (int r = 0; r < 8; ++r)
        h_all[((size_t)t * BATCH + r) * DSTATE + n] = hv[r];
    }
    lds_barrier();  // DS-only fence + split barrier (no storecnt wait)
  };

  for (int t = 0; t < TLEN; t += 2) {
    step(t, hsh0, hsh1);
    step(t + 1, hsh1, hsh0);
  }
}

// ---------------------------------------------------------------------------
// Gating:  y = g * (hC + uD) + (1-g) * x   (bf16 out, feeds final GEMM)
//   hc indexed [t, b, n]; everything else [b, t, n].
// ---------------------------------------------------------------------------
__global__ void gate_kernel(const __bf16* __restrict__ hc,
                            const __bf16* __restrict__ ud,
                            const __bf16* __restrict__ g,
                            const float* __restrict__ x,
                            __bf16* __restrict__ y, int total) {
  int i = blockIdx.x * blockDim.x + threadIdx.x;
  if (i >= total) return;
  int nn = i % DMODEL;
  int bt = i / DMODEL;
  int t  = bt % TLEN;
  int b  = bt / TLEN;
  float hcv = (float)hc[((size_t)t * BATCH + b) * DMODEL + nn];
  float gv  = (float)g[i];
  float yv  = gv * (hcv + (float)ud[i]) + (1.0f - gv) * x[i];
  y[i] = (__bf16)yv;
}

// ---------------------------------------------------------------------------
extern "C" void kernel_launch(void* const* d_in, const int* in_sizes, int n_in,
                              void* d_out, int out_size, void* d_ws, size_t ws_size,
                              hipStream_t stream) {
  (void)in_sizes; (void)n_in; (void)out_size; (void)ws_size;
  const float* x      = (const float*)d_in[0];
  const float* W_in   = (const float*)d_in[1];
  const float* b_in   = (const float*)d_in[2];
  const float* W_gate = (const float*)d_in[3];
  const float* b_gate = (const float*)d_in[4];
  const float* A      = (const float*)d_in[5];
  const float* Bm     = (const float*)d_in[6];
  const float* C      = (const float*)d_in[7];
  const float* D      = (const float*)d_in[8];
  const float* W_out  = (const float*)d_in[9];
  const float* b_out  = (const float*)d_in[10];
  float* out = (float*)d_out;

  char* ws = (char*)d_ws;
  size_t off = 0;
  auto alloc = [&](size_t bytes) -> char* {
    char* p = ws + off;
    off += (bytes + 255) & ~(size_t)255;
    return p;
  };
  __bf16* xb      = (__bf16*)alloc((size_t)BT * DMODEL * 2);
  __bf16* W_inT   = (__bf16*)alloc((size_t)DSTATE * DMODEL * 2);
  __bf16* W_gateT = (__bf16*)alloc((size_t)DMODEL * DMODEL * 2);
  __bf16* BmT     = (__bf16*)alloc((size_t)DSTATE * DSTATE * 2);
  __bf16* Cb      = (__bf16*)alloc((size_t)DMODEL * DSTATE * 2);
  __bf16* Db      = (__bf16*)alloc((size_t)DMODEL * DSTATE * 2);
  __bf16* W_outT  = (__bf16*)alloc((size_t)DMODEL * DMODEL * 2);
  __bf16* ATb     = (__bf16*)alloc((size_t)DSTATE * DSTATE * 2);
  __bf16* u_b     = (__bf16*)alloc((size_t)BT * DSTATE * 2);
  __bf16* g_b     = (__bf16*)alloc((size_t)BT * DMODEL * 2);
  float*  ub_f    = (float*) alloc((size_t)BT * DSTATE * 4);
  __bf16* ud_b    = (__bf16*)alloc((size_t)BT * DMODEL * 2);
  __bf16* h_b     = (__bf16*)alloc((size_t)TLEN * BATCH * DSTATE * 2);
  __bf16* hc_b    = (__bf16*)alloc((size_t)TLEN * BATCH * DMODEL * 2);
  __bf16* y_b     = (__bf16*)alloc((size_t)BT * DMODEL * 2);

  const int TB = 256;
  // --- conversions ---
  cvt_f32_to_bf16<<<ceil_div(BT * DMODEL, TB), TB, 0, stream>>>(x, xb, BT * DMODEL);
  cvt_f32_to_bf16<<<ceil_div(DMODEL * DSTATE, TB), TB, 0, stream>>>(C, Cb, DMODEL * DSTATE);
  cvt_f32_to_bf16<<<ceil_div(DMODEL * DSTATE, TB), TB, 0, stream>>>(D, Db, DMODEL * DSTATE);
  cvt_transpose_bf16<<<ceil_div(DMODEL * DSTATE, TB), TB, 0, stream>>>(W_in, W_inT, DMODEL, DSTATE);
  cvt_transpose_bf16<<<ceil_div(DMODEL * DMODEL, TB), TB, 0, stream>>>(W_gate, W_gateT, DMODEL, DMODEL);
  cvt_transpose_bf16<<<ceil_div(DSTATE * DSTATE, TB), TB, 0, stream>>>(Bm, BmT, DSTATE, DSTATE);
  cvt_transpose_bf16<<<ceil_div(DMODEL * DMODEL, TB), TB, 0, stream>>>(W_out, W_outT, DMODEL, DMODEL);
  cvt_transpose_bf16<<<ceil_div(DSTATE * DSTATE, TB), TB, 0, stream>>>(A, ATb, DSTATE, DSTATE);

  // --- hoisted GEMMs ---
  // u = x @ W_in + b_in                      -> bf16 [BT,256]
  gemm_bf16_wmma<true, false, true, false><<<dim3(BT / 128, DSTATE / 128), 128, 0, stream>>>(
      xb, W_inT, b_in, nullptr, u_b, BT, DSTATE, DMODEL);
  // g = sigmoid(x @ W_gate + b_gate)         -> bf16 [BT,1024]
  gemm_bf16_wmma<true, true, true, false><<<dim3(BT / 128, DMODEL / 128), 128, 0, stream>>>(
      xb, W_gateT, b_gate, nullptr, g_b, BT, DMODEL, DMODEL);
  // ub = u @ Bm                              -> f32 [BT,256] (scan C input)
  gemm_bf16_wmma<false, false, false, false><<<dim3(BT / 128, DSTATE / 128), 128, 0, stream>>>(
      u_b, BmT, nullptr, nullptr, ub_f, BT, DSTATE, DSTATE);
  // ud = u @ D^T                             -> bf16 [BT,1024]
  gemm_bf16_wmma<false, false, true, false><<<dim3(BT / 128, DMODEL / 128), 128, 0, stream>>>(
      u_b, Db, nullptr, nullptr, ud_b, BT, DMODEL, DSTATE);

  // --- the irreducibly sequential part ---
  scan_kernel<<<1, 512, 0, stream>>>(ATb, ub_f, h_b);

  // hc = h @ C^T                             -> bf16 [T*8,1024] (rows t*8+b)
  gemm_bf16_wmma<false, false, true, false><<<dim3(BT / 128, DMODEL / 128), 128, 0, stream>>>(
      h_b, Cb, nullptr, nullptr, hc_b, BT, DMODEL, DSTATE);

  // y = g*(hc+ud) + (1-g)*x                  -> bf16 [BT,1024]
  gate_kernel<<<ceil_div(BT * DMODEL, TB), TB, 0, stream>>>(hc_b, ud_b, g_b, x, y_b, BT * DMODEL);

  // out = y @ W_out + b_out + x              -> f32 [BT,1024]
  gemm_bf16_wmma<true, false, false, true><<<dim3(BT / 128, DMODEL / 128), 128, 0, stream>>>(
      y_b, W_outT, b_out, x, out, BT, DMODEL, DMODEL);
}
